// HierarchicalSkeletalEncoder_75539884802739
// MI455X (gfx1250) — matrix operand order repaired and checked by reading, not compiled, not run on hardware
//
#include <hip/hip_runtime.h>
#include <hip/hip_bf16.h>

typedef __attribute__((ext_vector_type(16))) _Float16 v16h;
typedef __attribute__((ext_vector_type(8)))  _Float16 v8h;
typedef __attribute__((ext_vector_type(8)))  float    v8f;

#define NJ    17
#define FS    72      // f16 row stride (halves); 144B rows stay 16B-aligned
#define POOLW 384

__constant__ int c_subj[NJ]   = {0,-1,-1,-1,-1,0,0,1,1,2,2,0,0,1,1,2,2};
__constant__ int c_subs[3][5] = {{0,5,6,11,12},{7,8,13,14,0},{9,10,15,16,0}};
__constant__ int c_subn[3]    = {5,4,4};

struct __align__(16) Smem {
  float    px[272];          // P.x per (sample,joint) row
  float    py[272];          // P.y
  float    sc[272];          // score
  _Float16 feat[272*FS];     // H*sc, f16 (WMMA A operand)
  _Float16 abuf[272*FS];     // feat @ (W1-W2)^T
  _Float16 bbuf[272*FS];     // feat @ W2^T
  float    pool[16*POOLW];   // pooled (16 samples x 384)
  _Float16 poolH[16*POOLW];  // f16 copy for final WMMA
  unsigned adj[NJ];          // per-level adjacency bitmasks
};

__global__ __launch_bounds__(128)
void hse_main(const float* __restrict__ kp, const float* __restrict__ scg,
              const float* __restrict__ Wl, const float* __restrict__ bl,
              const float* __restrict__ bedge, const float* __restrict__ bproj,
              const _Float16* __restrict__ wb1, const _Float16* __restrict__ wb2,
              const _Float16* __restrict__ wpj,
              float* __restrict__ out, int N)
{
  extern __shared__ char smraw[];
  Smem& S = *reinterpret_cast<Smem*>(smraw);
  const int tid  = threadIdx.x;
  const int lane = tid & 31;
  const int wave = tid >> 5;
  const int hi   = (lane >> 4) & 1;
  const int ln   = lane & 15;
  const int blk  = blockIdx.x;

  // ---- Phase 0: per-sample keypoint normalization -> P rows in LDS ----
  if (tid < 16) {
    int s = blk*16 + tid; if (s >= N) s = N - 1;
    const float* kpp = kp + (size_t)s * NJ * 2;
    float mnx=3.4e38f, mny=3.4e38f, mxx=-3.4e38f, mxy=-3.4e38f;
    for (int j = 0; j < NJ; ++j) {
      float x = kpp[2*j], y = kpp[2*j+1];
      mnx = fminf(mnx,x); mxx = fmaxf(mxx,x);
      mny = fminf(mny,y); mxy = fmaxf(mxy,y);
    }
    float ix = 1.0f/(mxx-mnx+1e-6f), iy = 1.0f/(mxy-mny+1e-6f);
    for (int j = 0; j < NJ; ++j) {
      int r = tid*NJ + j;
      S.px[r] = (kpp[2*j]   - mnx) * ix;
      S.py[r] = (kpp[2*j+1] - mny) * iy;
      S.sc[r] = scg[(size_t)s*NJ + j];
    }
  }
  __syncthreads();

  for (int lvl = 0; lvl < 3; ++lvl) {
    // adjacency bitmask for this level (read after next barrier)
    if (tid < NJ) {
      unsigned m = 0; int sa = c_subj[tid];
      for (int k = 0; k < NJ; ++k) {
        if (k == tid) continue;
        int sb = c_subj[k];
        bool e = (sa==lvl && sb==lvl) ||
                 (sa==lvl && (sb==lvl-1 || sb==lvl+1)) ||
                 (sb==lvl && (sa==lvl-1 || sa==lvl+1));
        if (e) m |= 1u << k;
      }
      S.adj[tid] = m;
    }
    const int cnt = c_subn[lvl];

    // ---- Phase 1: h = relu(P Wl^T + b); H = subset-sum - self; feat = H*sc;
    //              pooled mean_j H = (cnt-1)/17 * sum_k hs_k  (analytic) ----
    for (int item = tid; item < 16*64; item += 128) {
      int n = item >> 6, d = item & 63;
      float w0 = Wl[(lvl*64 + d)*3 + 0];
      float w1 = Wl[(lvl*64 + d)*3 + 1];
      float w2 = Wl[(lvl*64 + d)*3 + 2];
      float b  = bl[lvl*64 + d];
      float hk[5]; float sum = 0.f;
      #pragma unroll
      for (int idx = 0; idx < 5; ++idx) {
        float h = 0.f;
        if (idx < cnt) {
          int k = c_subs[lvl][idx];
          int r = n*NJ + k;
          h = fmaxf(S.px[r]*w0 + S.py[r]*w1 + S.sc[r]*w2 + b, 0.f) * S.sc[r];
        }
        hk[idx] = h; sum += h;
      }
      #pragma unroll
      for (int j = 0; j < NJ; ++j) S.feat[(n*NJ + j)*FS + d] = (_Float16)0.f;
      #pragma unroll
      for (int idx = 0; idx < 5; ++idx) {
        if (idx < cnt) {
          int j = c_subs[lvl][idx];
          S.feat[(n*NJ + j)*FS + d] = (_Float16)((sum - hk[idx]) * S.sc[n*NJ + j]);
        }
      }
      S.pool[n*POOLW + lvl*128 + d] = (float)(cnt - 1) * sum * (1.0f/17.0f);
    }
    __syncthreads();

    // ---- Phase 2: WMMA GEMMs: a = feat@(W1-W2)^T, b = feat@W2^T
    //      272 rows = 17 M-tiles, 4 col-tiles (one per wave), K=64 (2 steps)
    {
      const int ct = wave;
      const int e  = ct*16 + ln;           // output column (B operand column)
      v16h B10, B11, B20, B21;
      {
        const _Float16* p = wb1 + e*64 + hi*8;           // kt=0
        ((v8h*)&B10)[0] = *(const v8h*)(p);
        ((v8h*)&B10)[1] = *(const v8h*)(p + 16);
        p = wb1 + e*64 + 32 + hi*8;                      // kt=1
        ((v8h*)&B11)[0] = *(const v8h*)(p);
        ((v8h*)&B11)[1] = *(const v8h*)(p + 16);
        p = wb2 + e*64 + hi*8;
        ((v8h*)&B20)[0] = *(const v8h*)(p);
        ((v8h*)&B20)[1] = *(const v8h*)(p + 16);
        p = wb2 + e*64 + 32 + hi*8;
        ((v8h*)&B21)[0] = *(const v8h*)(p);
        ((v8h*)&B21)[1] = *(const v8h*)(p + 16);
      }
      for (int mt = 0; mt < 17; ++mt) {
        int row = mt*16 + ln;
        v16h A0, A1;                                     // ISA 16-bit A layout
        {
          const _Float16* p = S.feat + row*FS + hi*8;    // kt=0: K = hi*8+0..7
          ((v8h*)&A0)[0] = *(const v8h*)(p);
          ((v8h*)&A0)[1] = *(const v8h*)(p + 16);        // K = 16+hi*8+0..7
          p = S.feat + row*FS + 32 + hi*8;               // kt=1
          ((v8h*)&A1)[0] = *(const v8h*)(p);
          ((v8h*)&A1)[1] = *(const v8h*)(p + 16);
        }
        v8f accA = {}; v8f accB = {};
        accA = __builtin_amdgcn_wmma_f32_16x16x32_f16(false, A0, false, B10, (short)0, accA, false, false);
        accA = __builtin_amdgcn_wmma_f32_16x16x32_f16(false, A1, false, B11, (short)0, accA, false, false);
        accB = __builtin_amdgcn_wmma_f32_16x16x32_f16(false, A0, false, B20, (short)0, accB, false, false);
        accB = __builtin_amdgcn_wmma_f32_16x16x32_f16(false, A1, false, B21, (short)0, accB, false, false);
        #pragma unroll
        for (int r = 0; r < 8; ++r) {                    // C/D layout: M=r (+8 hi)
          int orow = mt*16 + (hi ? 8 + r : r);
          S.abuf[orow*FS + e] = (_Float16)accA[r];
          S.bbuf[orow*FS + e] = (_Float16)accB[r];
        }
      }
    }
    __syncthreads();

    // ---- Phase 3: msgs = relu(a[dst]+b[src]+be); Z = segmax; pooled Z mean ----
    for (int item = tid; item < 16*64; item += 128) {
      int n = item >> 6, d = item & 63;
      float be = bedge[d];
      float av[NJ], bv[NJ];
      #pragma unroll
      for (int j = 0; j < NJ; ++j) {
        av[j] = (float)S.abuf[(n*NJ + j)*FS + d];
        bv[j] = (float)S.bbuf[(n*NJ + j)*FS + d];
      }
      float acc = 0.f;
      #pragma unroll
      for (int j = 0; j < NJ; ++j) {
        unsigned m = S.adj[j];
        float mx = -3.4e38f;
        #pragma unroll
        for (int k = 0; k < NJ; ++k)
          if (m & (1u << k)) mx = fmaxf(mx, bv[k]);
        float z = fmaxf(av[j] + be + mx, 0.f);           // relu∘max == max∘relu
        acc += (m != 0u) ? z : 0.f;
      }
      S.pool[n*POOLW + lvl*128 + 64 + d] = acc * (1.0f/17.0f);
    }
    __syncthreads();
  }

  // ---- Final projection: out(16x128) = pooled(16x384) @ Wproj^T + bproj ----
  for (int i = tid; i < 16*POOLW; i += 128) S.poolH[i] = (_Float16)S.pool[i];
  __syncthreads();

  for (int c = wave*2; c < wave*2 + 2; ++c) {            // 8 col-tiles / 4 waves
    int col = c*16 + ln;
    v8f acc = {};
    for (int kt = 0; kt < 12; ++kt) {                    // K = 384
      v16h A, B;
      const _Float16* p = S.poolH + ln*POOLW + kt*32 + hi*8;
      ((v8h*)&A)[0] = *(const v8h*)(p);
      ((v8h*)&A)[1] = *(const v8h*)(p + 16);
      const _Float16* q = wpj + col*POOLW + kt*32 + hi*8;
      ((v8h*)&B)[0] = *(const v8h*)(q);
      ((v8h*)&B)[1] = *(const v8h*)(q + 16);
      acc = __builtin_amdgcn_wmma_f32_16x16x32_f16(false, A, false, B, (short)0, acc, false, false);
    }
    float bp = bproj[col];
    #pragma unroll
    for (int r = 0; r < 8; ++r) {
      int srow = hi ? 8 + r : r;
      int s = blk*16 + srow;
      if (s < N) out[(size_t)s*128 + col] = acc[r] + bp;
    }
  }
}

// Prep: f16 weight copies in B-friendly layouts ([col][K] row-major => per-lane
// contiguous K runs => b128 loads). L2-resident, shared by all blocks.
__global__ void hse_prep(const float* __restrict__ Wedge, const float* __restrict__ Wproj,
                         _Float16* __restrict__ wb1, _Float16* __restrict__ wb2,
                         _Float16* __restrict__ wpj)
{
  int tid = blockIdx.x * blockDim.x + threadIdx.x;
  int nt  = gridDim.x * blockDim.x;
  for (int i = tid; i < 64*64; i += nt) {
    int e = i >> 6, d = i & 63;
    float w1 = Wedge[e*128 + d], w2 = Wedge[e*128 + 64 + d];
    wb1[i] = (_Float16)(w1 - w2);
    wb2[i] = (_Float16)w2;
  }
  for (int i = tid; i < 128*384; i += nt) wpj[i] = (_Float16)Wproj[i];
}

extern "C" void kernel_launch(void* const* d_in, const int* in_sizes, int n_in,
                              void* d_out, int out_size, void* d_ws, size_t ws_size,
                              hipStream_t stream)
{
  (void)n_in; (void)out_size; (void)ws_size;
  const float* kp  = (const float*)d_in[0];
  const float* scg = (const float*)d_in[1];
  const float* Wl  = (const float*)d_in[2];
  const float* bl  = (const float*)d_in[3];
  const float* Wed = (const float*)d_in[4];
  const float* bed = (const float*)d_in[5];
  const float* Wpj = (const float*)d_in[6];
  const float* bpj = (const float*)d_in[7];
  float* out = (float*)d_out;

  int N = in_sizes[0] / (NJ * 2);

  _Float16* wb1 = (_Float16*)d_ws;
  _Float16* wb2 = wb1 + 64*64;
  _Float16* wpj = wb2 + 64*64;

  hse_prep<<<8, 256, 0, stream>>>(Wed, Wpj, wb1, wb2, wpj);

  int blocks = (N + 15) / 16;
  hse_main<<<blocks, 128, sizeof(Smem), stream>>>(kp, scg, Wl, bl, bed, bpj,
                                                  wb1, wb2, wpj, out, N);
}
